// NeuralKalmanFilter_29265907155787
// MI455X (gfx1250) — matrix operand order repaired and checked by reading, not compiled, not run on hardware
//
#include <hip/hip_runtime.h>
#include <hip/hip_bf16.h>

#define LATENT 2048
#define OBS    1024
#define CTRL   512
#define SEQ    4096

#define NBLK 128
#define NTHR 256   // 8 waves of 32 -> 1024 waves total

typedef __attribute__((ext_vector_type(16))) _Float16 v16h;
typedef __attribute__((ext_vector_type(8)))  _Float16 v8h;
typedef __attribute__((ext_vector_type(8)))  float    v8f;

// ---------------- grid-wide barrier (sense-reversing, L2 atomics) ----------
__device__ inline void gsync(unsigned* cnt, unsigned* gen, unsigned nblk) {
  __syncthreads();
  if (threadIdx.x == 0) {
    __threadfence();
    volatile unsigned* vgen = (volatile unsigned*)gen;
    unsigned g = *vgen;
    unsigned arrived = atomicAdd(cnt, 1u);
    if (arrived == nblk - 1u) {
      *(volatile unsigned*)cnt = 0u;   // safe: all others spin on gen
      __threadfence();
      atomicAdd(gen, 1u);
    } else {
      while (*vgen == g) { __builtin_amdgcn_s_sleep(4); }
    }
    __threadfence();
  }
  __syncthreads();
}

__device__ inline float wave_sum(float v) {
  #pragma unroll
  for (int off = 16; off >= 1; off >>= 1)
    v += __shfl_xor(v, off, 32);
  return v;
}

// ---------------- workspace layout (shared across kernels) -----------------
// [0..63]  : unsigned cnt, gen (+ padding)
// floats   : z(2048) fzprev(2048) fu(512) drive(2048) fz(2048) ex(1024) gpart(16*2048)
// halves   : Wh (OBS x LATENT, row-major)  Ht (SEQ x LATENT, row-major == B^T)
struct WS {
  unsigned *cnt, *gen;
  float *z, *fzprev, *fu, *drive, *fz, *ex, *gpart;
  _Float16 *Wh, *Ht;
};
__device__ __host__ inline WS ws_layout(void* d_ws) {
  WS w;
  float* f = (float*)d_ws;
  w.cnt = (unsigned*)f; w.gen = (unsigned*)f + 1;
  w.z      = f + 64;
  w.fzprev = w.z      + LATENT;
  w.fu     = w.fzprev + LATENT;
  w.drive  = w.fu     + CTRL;
  w.fz     = w.drive  + LATENT;
  w.ex     = w.fz     + LATENT;
  w.gpart  = w.ex     + OBS;
  w.Wh = (_Float16*)(w.gpart + 16 * LATENT);
  w.Ht = w.Wh + (size_t)OBS * LATENT;
  return w;
}

__global__ void nkf_init(float* wsf) {
  ((unsigned*)wsf)[0] = 0u;
  ((unsigned*)wsf)[1] = 0u;
}

// ---------------- persistent sequential kernel -----------------------------
__global__ __launch_bounds__(NTHR)
void nkf_seq(const float* __restrict__ inputs,    // (OBS, SEQ)
             const float* __restrict__ controls,  // (CTRL, SEQ)
             const float* __restrict__ Wr,        // (LATENT, LATENT)
             const float* __restrict__ Win,       // (LATENT, CTRL)
             const float* __restrict__ Wout,      // (OBS, LATENT)
             const int*   __restrict__ p_iters,
             const float* __restrict__ p_lr,
             float* __restrict__ zs_out,          // (LATENT, SEQ)
             float* __restrict__ wsf) {
  WS w = ws_layout(wsf);
  const int tid  = blockIdx.x * NTHR + threadIdx.x;   // 0..32767
  const int lane = threadIdx.x & 31;
  const int wid  = tid >> 5;                          // 0..1023
  const int   niter = *p_iters;
  const float lr    = *p_lr;

  // phase 0: z = 0, Wout -> f16 (done once; all weights then live in L2)
  if (tid < LATENT) w.z[tid] = 0.0f;
  for (size_t i = tid; i < (size_t)OBS * LATENT; i += (size_t)NBLK * NTHR)
    w.Wh[i] = (_Float16)Wout[i];
  gsync(w.cnt, w.gen, NBLK);

  for (int t = 0; t < SEQ; ++t) {
    // phase 1: fzprev = tanh(z), fu = tanh(u_t)
    if (tid < LATENT) {
      w.fzprev[tid] = tanhf(w.z[tid]);
    } else if (tid < LATENT + CTRL) {
      int j = tid - LATENT;
      w.fu[j] = tanhf(controls[(size_t)j * SEQ + t]);
    }
    gsync(w.cnt, w.gen, NBLK);

    // phase 2: drive = Wr @ fzprev + Win @ fu ;
    // stage tanh(drive) into Ht TRANSPOSED: Ht[t][r]  (SEQ x LATENT)
    #pragma unroll
    for (int rr = 0; rr < 2; ++rr) {
      int r = wid + rr * 1024;
      float acc = 0.0f;
      const float* wrow = Wr + (size_t)r * LATENT;
      for (int k = lane; k < LATENT; k += 32) acc += wrow[k] * w.fzprev[k];
      const float* irow = Win + (size_t)r * CTRL;
      for (int k = lane; k < CTRL; k += 32)   acc += irow[k] * w.fu[k];
      acc = wave_sum(acc);
      if (lane == 0) {
        w.drive[r] = acc;
        w.Ht[(size_t)t * LATENT + r] = (_Float16)tanhf(acc);
      }
    }
    gsync(w.cnt, w.gen, NBLK);

    // inference loop
    for (int it = 0; it < niter; ++it) {
      // 3a: fz = tanh(z)
      if (tid < LATENT) w.fz[tid] = tanhf(w.z[tid]);
      gsync(w.cnt, w.gen, NBLK);

      // 3b: ex = x - Wout @ fz   (one row per wave, 1024 waves == OBS rows)
      {
        int i = wid;  // < OBS
        float acc = 0.0f;
        const float* wrow = Wout + (size_t)i * LATENT;
        for (int k = lane; k < LATENT; k += 32) acc += wrow[k] * w.fz[k];
        acc = wave_sum(acc);
        if (lane == 0) w.ex[i] = inputs[(size_t)i * SEQ + t] - acc;
      }
      gsync(w.cnt, w.gen, NBLK);

      // 3c: gpart[kc][i] = partial of (Wout^T @ ex); coalesced column reads
      {
        int ig = wid >> 4;         // 0..63 -> 32 outputs each
        int kc = wid & 15;         // 16 k-chunks of 64
        int i  = ig * 32 + lane;   // 0..2047
        int kbeg = kc * 64;
        float acc = 0.0f;
        for (int k = kbeg; k < kbeg + 64; ++k)
          acc += Wout[(size_t)k * LATENT + i] * w.ex[k];
        w.gpart[(size_t)kc * LATENT + i] = acc;
      }
      gsync(w.cnt, w.gen, NBLK);

      // 3d: g = sum partials ; z -= lr * ((z - drive) - (1 - fz^2) * g)
      if (tid < LATENT) {
        float g = 0.0f;
        #pragma unroll
        for (int c2 = 0; c2 < 16; ++c2) g += w.gpart[(size_t)c2 * LATENT + tid];
        float f  = w.fz[tid];
        float zi = w.z[tid];
        float dz = (zi - w.drive[tid]) - (1.0f - f * f) * g;
        w.z[tid] = zi - lr * dz;
      }
      gsync(w.cnt, w.gen, NBLK);
    }

    // emit zs column t (z is synced by the last barrier; pure read of z)
    if (tid < LATENT) zs_out[(size_t)tid * SEQ + t] = w.z[tid];
  }
}

// ---------------- pred_xs = Wout(f16) @ tanh(Zproj)(f16), WMMA f32 acc -----
// C (OBS x SEQ) tiled 16x16 per wave; K = LATENT in steps of 64 with two
// independent accumulator chains (hides XDL latency; deterministic final add).
// B is stored transposed (Ht: SEQ x LATENT), so both fragments load as two
// aligned 16-byte vectors per lane.
__global__ __launch_bounds__(256)
void nkf_predx_gemm(const _Float16* __restrict__ Wh,   // (OBS, LATENT) row-major
                    const _Float16* __restrict__ Ht,   // (SEQ, LATENT) row-major (B^T)
                    float* __restrict__ out) {         // (OBS, SEQ)
  const int lane = threadIdx.x & 31;
  const int wid  = (blockIdx.x * blockDim.x + threadIdx.x) >> 5;  // 0..16383
  const int m0 = (wid >> 8) << 4;    // 64 M-tiles
  const int n0 = (wid & 255) << 4;   // 256 N-tiles
  const int rc   = lane & 15;                 // A row / B col within tile
  const int akb  = (lane < 16) ? 0 : 8;       // A: lanes 16-31 hold K=8..15 / 24..31
  const int bkof = (lane < 16) ? 0 : 8;       // B: lanes 16-31 hold K=8..15 / 24..31

  v8f c0 = {};
  v8f c1 = {};
  const _Float16* arow = Wh + (size_t)(m0 + rc) * LATENT;
  const _Float16* brow = Ht + (size_t)(n0 + rc) * LATENT;

  for (int k0 = 0; k0 < LATENT; k0 += 64) {
    // chain 0: K = [k0, k0+32)
    {
      v8h alo = *(const v8h*)(arow + k0 + akb);
      v8h ahi = *(const v8h*)(arow + k0 + 16 + akb);
      v16h a = __builtin_shufflevector(alo, ahi,
               0,1,2,3,4,5,6,7, 8,9,10,11,12,13,14,15);
      v8h blo = *(const v8h*)(brow + k0 + bkof);
      v8h bhi = *(const v8h*)(brow + k0 + 16 + bkof);
      v16h b = __builtin_shufflevector(blo, bhi,
               0,1,2,3,4,5,6,7, 8,9,10,11,12,13,14,15);
      c0 = __builtin_amdgcn_wmma_f32_16x16x32_f16(
              false, a, false, b, (short)0, c0, false, false);
    }
    // chain 1: K = [k0+32, k0+64)
    {
      int k1 = k0 + 32;
      v8h alo = *(const v8h*)(arow + k1 + akb);
      v8h ahi = *(const v8h*)(arow + k1 + 16 + akb);
      v16h a = __builtin_shufflevector(alo, ahi,
               0,1,2,3,4,5,6,7, 8,9,10,11,12,13,14,15);
      v8h blo = *(const v8h*)(brow + k1 + bkof);
      v8h bhi = *(const v8h*)(brow + k1 + 16 + bkof);
      v16h b = __builtin_shufflevector(blo, bhi,
               0,1,2,3,4,5,6,7, 8,9,10,11,12,13,14,15);
      c1 = __builtin_amdgcn_wmma_f32_16x16x32_f16(
              false, a, false, b, (short)0, c1, false, false);
    }
  }
  v8f c = c0 + c1;

  // C/D layout: VGPR r -> M=r (lanes 0-15), M=r+8 (lanes 16-31); N = lane&15
  #pragma unroll
  for (int r = 0; r < 8; ++r) {
    int m = m0 + r + ((lane >> 4) << 3);
    out[(size_t)m * SEQ + (n0 + rc)] = c[r];
  }
}

extern "C" void kernel_launch(void* const* d_in, const int* in_sizes, int n_in,
                              void* d_out, int out_size, void* d_ws, size_t ws_size,
                              hipStream_t stream) {
  const float* inputs   = (const float*)d_in[0];  // (OBS, SEQ)
  const float* controls = (const float*)d_in[1];  // (CTRL, SEQ)
  const float* Wr       = (const float*)d_in[2];  // (LATENT, LATENT)
  const float* Win      = (const float*)d_in[3];  // (LATENT, CTRL)
  const float* Wout     = (const float*)d_in[4];  // (OBS, LATENT)
  const int*   p_iters  = (const int*)  d_in[5];  // scalar
  const float* p_lr     = (const float*)d_in[6];  // scalar

  float* zs_out    = (float*)d_out;                          // (LATENT, SEQ)
  float* predx_out = (float*)d_out + (size_t)LATENT * SEQ;   // (OBS, SEQ)
  float* wsf       = (float*)d_ws;
  WS w = ws_layout(d_ws);

  nkf_init<<<1, 1, 0, stream>>>(wsf);

  nkf_seq<<<NBLK, NTHR, 0, stream>>>(inputs, controls, Wr, Win, Wout,
                                     p_iters, p_lr, zs_out, wsf);

  // 16384 output tiles of 16x16, 8 waves per block -> 2048 blocks
  nkf_predx_gemm<<<2048, 256, 0, stream>>>(w.Wh, w.Ht, predx_out);

  (void)in_sizes; (void)n_in; (void)out_size; (void)ws_size;
}
